// GraphAttentionLayer_81827716923894
// MI455X (gfx1250) — matrix operand order repaired and checked by reading, not compiled, not run on hardware
//
#include <hip/hip_runtime.h>
#include <hip/hip_bf16.h>

typedef __attribute__((ext_vector_type(16))) _Float16 v16h;
typedef __attribute__((ext_vector_type(8)))  _Float16 v8h;
typedef __attribute__((ext_vector_type(8)))  float    v8f;
typedef __attribute__((ext_vector_type(4)))  float    v4f;

#define BB     8
#define NN     2048
#define FIN    256
#define OUTD   256
#define NH     4
#define DH     64
#define BIGNEG (-3.0e38f)
#define SM_SHIFT 8.0f
#define SM_CLAMP 19.0f   // exp(19-8) = 59874 < f16 max 65504

// ---------------------------------------------------------------------------
// cross-lane sum within 16-lane row groups (wave32) -- used only outside loops
// ---------------------------------------------------------------------------
__device__ __forceinline__ float rowred_sum(float x) {
#pragma unroll
  for (int m = 1; m < 16; m <<= 1) x += __shfl_xor(x, m, 16);
  return x;
}

// ---------------------------------------------------------------------------
// Kernel 0a: swizzle weights (fp32 row-major -> f16 B-fragment layout)
// dst[w][(kc*16+cc)*32 + lane][i] = W[kc*32 + ((i&8)<<1) + (lane<16?0:8) + (i&7)][cc*16 + (lane&15)]
// ---------------------------------------------------------------------------
__global__ void gat_swz_w(const float* __restrict__ wq, const float* __restrict__ wk,
                          const float* __restrict__ wv, const float* __restrict__ wo,
                          _Float16* __restrict__ dst) {
  int idx  = blockIdx.x * blockDim.x + threadIdx.x;   // 262144 total
  int i    = idx & 15;
  int lane = (idx >> 4) & 31;
  int cc   = (idx >> 9) & 15;
  int kc   = (idx >> 13) & 7;
  int w    = idx >> 16;
  const float* src = (w == 0) ? wq : (w == 1) ? wk : (w == 2) ? wv : wo;
  int krow = kc * 32 + ((i & 8) << 1) + ((lane < 16) ? 0 : 8) + (i & 7);
  int col  = cc * 16 + (lane & 15);
  dst[(size_t)w * 65536 + ((size_t)((kc * 16 + cc) * 32 + lane)) * 16 + i] =
      (_Float16)src[(size_t)krow * OUTD + col];
}

// ---------------------------------------------------------------------------
// Kernel 0b: repack adjacency: adjB[n>>4][m][n&15] = (f16)adj[n][m]
// lets each attention lane fetch its 8 row-values with one 16B load
// ---------------------------------------------------------------------------
__global__ void gat_swz_adj(const float* __restrict__ adj, _Float16* __restrict__ adjB) {
  size_t idx = (size_t)blockIdx.x * blockDim.x + threadIdx.x;  // 4194304 total
  int r  = idx & 15;
  int m  = (int)((idx >> 4) & (NN - 1));
  int nt = (int)(idx >> 15);
  adjB[idx] = (_Float16)adj[(size_t)(nt * 16 + r) * NN + m];
}

// ---------------------------------------------------------------------------
// Kernel 1: fused QKV projection.  One wave per 16-row tile of (B*N).
// Q -> [bh][n][64] (pre-scaled by 1/sqrt(DH)), K -> [bh][n][64], V -> [bh][64][n]
// ---------------------------------------------------------------------------
__global__ void gat_qkv(const float* __restrict__ x, const _Float16* __restrict__ wswz,
                        _Float16* __restrict__ q, _Float16* __restrict__ k,
                        _Float16* __restrict__ vt) {
  int wave  = threadIdx.x >> 5;
  int lane  = threadIdx.x & 31;
  int tile  = blockIdx.x * 4 + wave;            // 1024 tiles
  int col16 = lane & 15;
  int hi    = lane >> 4;
  int o     = hi ? 8 : 0;
  int base  = tile * 16;
  int bb    = base >> 11;
  int nloc  = base & (NN - 1);

  const float* xr = x + (size_t)(base + col16) * FIN;
  v16h a[8];
#pragma unroll
  for (int kc = 0; kc < 8; ++kc) {
    const float* p0 = xr + kc * 32 + o;
    const float* p1 = xr + kc * 32 + 16 + o;
    v4f f0 = *(const v4f*)p0, f1 = *(const v4f*)(p0 + 4);
    v4f g0 = *(const v4f*)p1, g1 = *(const v4f*)(p1 + 4);
    v16h av;
#pragma unroll
    for (int i = 0; i < 4; ++i) {
      av[i]      = (_Float16)f0[i];
      av[4 + i]  = (_Float16)f1[i];
      av[8 + i]  = (_Float16)g0[i];
      av[12 + i] = (_Float16)g1[i];
    }
    a[kc] = av;
  }

#pragma unroll
  for (int w = 0; w < 3; ++w) {
    const _Float16* wsw = wswz + (size_t)w * 65536;
#pragma unroll 4
    for (int cc = 0; cc < 16; ++cc) {
      v8f acc = {};
#pragma unroll
      for (int kc = 0; kc < 8; ++kc) {
        v16h bf = *(const v16h*)(wsw + ((size_t)((kc * 16 + cc) * 32 + lane)) * 16);
        acc = __builtin_amdgcn_wmma_f32_16x16x32_f16(false, a[kc], false, bf,
                                                     (short)0, acc, false, false);
      }
      int col = cc * 16 + col16;
      int hh  = col >> 6;
      int d   = col & 63;
      size_t bh = (size_t)bb * NH + hh;
#pragma unroll
      for (int v = 0; v < 8; ++v) {
        int n = nloc + v + 8 * hi;
        float val = acc[v];
        if (w == 0)      q[(bh * NN + n) * DH + d] = (_Float16)(val * 0.125f);
        else if (w == 1) k[(bh * NN + n) * DH + d] = (_Float16)val;
        else             vt[(bh * DH + d) * NN + n] = (_Float16)val;
      }
    }
  }
}

// ---------------------------------------------------------------------------
// Kernel 2: masked graph attention, static-shift softmax (no online rescale).
// One wave per (b, h, 32-row block): two 16-row subtiles share K/V fragments.
// ---------------------------------------------------------------------------
__global__ void gat_attn(const _Float16* __restrict__ q, const _Float16* __restrict__ k,
                         const _Float16* __restrict__ vt, const _Float16* __restrict__ adjB,
                         _Float16* __restrict__ o16) {
  __shared__ __align__(32) _Float16 pbuf[4][2][16 * 32];

  int wave  = threadIdx.x >> 5;
  int lane  = threadIdx.x & 31;
  int wid   = blockIdx.x * 4 + wave;            // 2048 waves
  int col16 = lane & 15;
  int hi    = lane >> 4;
  int o     = hi ? 8 : 0;
  int kb    = hi ? 16 : 0;

  int bh     = wid >> 6;                        // 32 (b,h) pairs
  int ntile2 = wid & 63;                        // 64 blocks of 32 rows
  int bb     = bh >> 2;
  int hh     = bh & 3;
  int n0     = ntile2 * 32;

  // Q A-fragments: [subtile r][d-chunk db]
  v16h qa[2][2];
#pragma unroll
  for (int r = 0; r < 2; ++r) {
    const _Float16* qr = q + ((size_t)bh * NN + n0 + r * 16 + col16) * DH;
#pragma unroll
    for (int db = 0; db < 2; ++db) {
      v8h l0 = *(const v8h*)(qr + db * 32 + 2 * o);
      v8h l1 = *(const v8h*)(qr + db * 32 + 2 * o + 8);
      v16h av;
#pragma unroll
      for (int i = 0; i < 8; ++i) { av[i] = l0[i]; av[8 + i] = l1[i]; }
      qa[r][db] = av;
    }
  }

  const _Float16* adjT0 = adjB + (size_t)(ntile2 * 2)     * (NN * 16) + 8 * hi;
  const _Float16* adjT1 = adjB + (size_t)(ntile2 * 2 + 1) * (NN * 16) + 8 * hi;

  float zacc[2][8];
  v8f   U[2][4];
#pragma unroll
  for (int r = 0; r < 2; ++r) {
#pragma unroll
    for (int v = 0; v < 8; ++v) zacc[r][v] = 0.0f;
#pragma unroll
    for (int dc = 0; dc < 4; ++dc) U[r][dc] = {};
  }

  for (int m0 = 0; m0 < NN; m0 += 32) {
    // ---- K B-fragments, shared by both row-subtiles ----
    v16h kf[2][2];
#pragma unroll
    for (int t = 0; t < 2; ++t)
#pragma unroll
      for (int db = 0; db < 2; ++db)
        kf[t][db] = *(const v16h*)(k + ((size_t)bh * NN + m0 + 16 * t + col16) * DH +
                                   db * 32 + kb);

#pragma unroll
    for (int r = 0; r < 2; ++r) {
      // scores (scale folded into Q)
      v8f S[2];
#pragma unroll
      for (int t = 0; t < 2; ++t) {
        v8f s = {};
        s = __builtin_amdgcn_wmma_f32_16x16x32_f16(false, qa[r][0], false, kf[t][0],
                                                   (short)0, s, false, false);
        s = __builtin_amdgcn_wmma_f32_16x16x32_f16(false, qa[r][1], false, kf[t][1],
                                                   (short)0, s, false, false);
        S[t] = s;
      }

      // adjacency: one 16B load per subtile-column-group (blocked f16 layout)
      const _Float16* at = (r == 0) ? adjT0 : adjT1;
      v8h a0 = *(const v8h*)(at + (size_t)(m0 + col16) * 16);
      v8h a1 = *(const v8h*)(at + (size_t)(m0 + 16 + col16) * 16);

      _Float16* pb = &pbuf[wave][r][0];
#pragma unroll
      for (int v = 0; v < 8; ++v) {
        float af0 = (float)a0[v], af1 = (float)a1[v];
        float s0 = (af0 == 0.0f) ? BIGNEG : S[0][v];
        float s1 = (af1 == 0.0f) ? BIGNEG : S[1][v];
        // static-shift softmax numerator: exact 0 for masked (exp(-3e38) -> 0)
        float p0 = __expf(fminf(s0, SM_CLAMP) - SM_SHIFT);
        float p1 = __expf(fminf(s1, SM_CLAMP) - SM_SHIFT);
        zacc[r][v] += p0 + p1;
        pb[(v + 8 * hi) * 32 + col16]      = (_Float16)(p0 * af0);
        pb[(v + 8 * hi) * 32 + 16 + col16] = (_Float16)(p1 * af1);
      }
    }

    // same-wave LDS ops are in-order; keep the compiler from hoisting the loads
    asm volatile("s_wait_dscnt 0" ::: "memory");

    // ---- P A-fragments (row-major LDS tile -> A layout) ----
    v16h pa[2];
#pragma unroll
    for (int r = 0; r < 2; ++r) {
      const _Float16* pr = &pbuf[wave][r][(size_t)col16 * 32];
      v8h pl0 = *(const v8h*)(pr + 2 * o);
      v8h pl1 = *(const v8h*)(pr + 2 * o + 8);
      v16h av;
#pragma unroll
      for (int i = 0; i < 8; ++i) { av[i] = pl0[i]; av[8 + i] = pl1[i]; }
      pa[r] = av;
    }

    // ---- V B-fragments, shared by both subtiles; U += P @ V ----
#pragma unroll
    for (int dc = 0; dc < 4; ++dc) {
      v16h vf = *(const v16h*)(vt + ((size_t)bh * DH + dc * 16 + col16) * NN + m0 + kb);
      U[0][dc] = __builtin_amdgcn_wmma_f32_16x16x32_f16(false, pa[0], false, vf,
                                                        (short)0, U[0][dc], false, false);
      U[1][dc] = __builtin_amdgcn_wmma_f32_16x16x32_f16(false, pa[1], false, vf,
                                                        (short)0, U[1][dc], false, false);
    }
  }

  // ---- single deferred Z reduction, normalize, store O ----
#pragma unroll
  for (int r = 0; r < 2; ++r) {
    float inv[8];
#pragma unroll
    for (int v = 0; v < 8; ++v) inv[v] = 1.0f / rowred_sum(zacc[r][v]);  // self-loop => Z>0
#pragma unroll
    for (int dc = 0; dc < 4; ++dc)
#pragma unroll
      for (int v = 0; v < 8; ++v) {
        int n = n0 + r * 16 + v + 8 * hi;
        o16[((size_t)bb * NN + n) * OUTD + hh * DH + dc * 16 + col16] =
            (_Float16)(U[r][dc][v] * inv[v]);
      }
  }
}

// ---------------------------------------------------------------------------
// Kernel 3: y = O @ Wo + bo + x (residual), then LayerNorm(gamma, beta).
// ---------------------------------------------------------------------------
__global__ void gat_outln(const _Float16* __restrict__ o16, const _Float16* __restrict__ wo_s,
                          const float* __restrict__ x, const float* __restrict__ bo,
                          const float* __restrict__ gamma, const float* __restrict__ beta,
                          float* __restrict__ out) {
  int wave  = threadIdx.x >> 5;
  int lane  = threadIdx.x & 31;
  int tile  = blockIdx.x * 4 + wave;
  int col16 = lane & 15;
  int hi    = lane >> 4;
  int o     = hi ? 8 : 0;
  int base  = tile * 16;

  const _Float16* orow = o16 + (size_t)(base + col16) * OUTD;
  v16h a[8];
#pragma unroll
  for (int kc = 0; kc < 8; ++kc) {
    v8h l0 = *(const v8h*)(orow + kc * 32 + o);
    v8h l1 = *(const v8h*)(orow + kc * 32 + 16 + o);
    v16h av;
#pragma unroll
    for (int i = 0; i < 8; ++i) { av[i] = l0[i]; av[8 + i] = l1[i]; }
    a[kc] = av;
  }

  v8f y[16];
#pragma unroll
  for (int cc = 0; cc < 16; ++cc) {
    v8f acc = {};
#pragma unroll
    for (int kc = 0; kc < 8; ++kc) {
      v16h bf = *(const v16h*)(wo_s + ((size_t)((kc * 16 + cc) * 32 + lane)) * 16);
      acc = __builtin_amdgcn_wmma_f32_16x16x32_f16(false, a[kc], false, bf,
                                                   (short)0, acc, false, false);
    }
    int col  = cc * 16 + col16;
    float bc = bo[col];
#pragma unroll
    for (int v = 0; v < 8; ++v)
      acc[v] += bc + x[(size_t)(base + v + 8 * hi) * FIN + col];
    y[cc] = acc;
  }

  v8f s1 = {}, s2 = {};
#pragma unroll
  for (int cc = 0; cc < 16; ++cc) { s1 += y[cc]; s2 += y[cc] * y[cc]; }
  float mu[8], rs[8];
#pragma unroll
  for (int v = 0; v < 8; ++v) {
    float m = rowred_sum(s1[v]) * (1.0f / 256.0f);
    float qv = rowred_sum(s2[v]) * (1.0f / 256.0f);
    mu[v] = m;
    rs[v] = rsqrtf(qv - m * m + 1e-5f);
  }

#pragma unroll
  for (int cc = 0; cc < 16; ++cc) {
    int col = cc * 16 + col16;
    float g = gamma[col], be = beta[col];
#pragma unroll
    for (int v = 0; v < 8; ++v)
      out[(size_t)(base + v + 8 * hi) * OUTD + col] =
          (y[cc][v] - mu[v]) * rs[v] * g + be;
  }
}

// ---------------------------------------------------------------------------
extern "C" void kernel_launch(void* const* d_in, const int* in_sizes, int n_in,
                              void* d_out, int out_size, void* d_ws, size_t ws_size,
                              hipStream_t stream) {
  const float* x     = (const float*)d_in[0];
  const float* adj   = (const float*)d_in[1];
  const float* Wq    = (const float*)d_in[2];
  const float* Wk    = (const float*)d_in[3];
  const float* Wv    = (const float*)d_in[4];
  const float* Wo    = (const float*)d_in[5];
  const float* bo    = (const float*)d_in[6];
  const float* gamma = (const float*)d_in[7];
  const float* beta  = (const float*)d_in[8];
  float* out = (float*)d_out;

  _Float16* ws16 = (_Float16*)d_ws;
  _Float16* wswz = ws16;                       //  4 * 65536 f16  (0.5 MB)
  _Float16* Q    = ws16 + 262144;              //  8 MB
  _Float16* K    = Q + 4194304;                //  8 MB
  _Float16* VT   = K + 4194304;                //  8 MB (transposed)
  _Float16* O    = VT + 4194304;               //  8 MB
  _Float16* ADJB = O + 4194304;                //  8 MB (blocked f16 adjacency)

  gat_swz_w  <<<1024, 256, 0, stream>>>(Wq, Wk, Wv, Wo, wswz);
  gat_swz_adj<<<16384, 256, 0, stream>>>(adj, ADJB);
  gat_qkv    <<<256, 128, 0, stream>>>(x, wswz, Q, K, VT);
  gat_attn   <<<512, 128, 0, stream>>>(Q, K, VT, ADJB, O);
  gat_outln  <<<256, 128, 0, stream>>>(O, wswz + 3 * 65536, x, bo, gamma, beta, out);
}
